// MiniBatchDiscriminator_62869731279616
// MI455X (gfx1250) — compile-verified, matrix operand order checked by reading, not run on hardware
//
#include <hip/hip_runtime.h>
#include <hip/hip_bf16.h>
#include <stdint.h>

#define BATCH 256
#define ADIM  1024
#define BDIM  64
#define CDIM  32
#define BC    (BDIM * CDIM)   // 2048
#define OUTW  (ADIM + BDIM)   // 1088

typedef __attribute__((ext_vector_type(16))) __bf16          v16bf;
typedef __attribute__((ext_vector_type(8)))  float           v8f;
typedef __attribute__((ext_vector_type(4)))  unsigned int    v4u;
typedef __attribute__((ext_vector_type(4)))  unsigned int    u32x4;
typedef __attribute__((ext_vector_type(8)))  int             i32x8;
typedef __attribute__((ext_vector_type(4)))  int             i32x4;

union FragU { v4u q[2]; v16bf b; };   // trivial types only -> OK in union

// round-to-nearest-even fp32 -> bf16 (manual, avoids libcalls)
__device__ __forceinline__ unsigned f2bf(float f) {
    unsigned u = __float_as_uint(f);
    u += 0x7FFFu + ((u >> 16) & 1u);
    return u >> 16;
}
__device__ __forceinline__ float bf2f(unsigned h) {
    return __uint_as_float(h << 16);
}

#if __has_builtin(__builtin_amdgcn_tensor_load_to_lds)
#define MBD_USE_TDM 1
#endif

#ifdef MBD_USE_TDM
// Build D# group 0: count=1, lds_addr, 57-bit global_addr, type=2.
__device__ __forceinline__ u32x4 tdm_g0(unsigned lds_addr, unsigned long long gaddr) {
    u32x4 g;
    g[0] = 1u;                                                   // count = 1
    g[1] = lds_addr;                                             // LDS byte addr
    g[2] = (unsigned)(gaddr & 0xFFFFFFFFull);                    // global[31:0]
    g[3] = (unsigned)((gaddr >> 32) & 0x01FFFFFFull) | 0x80000000u; // [56:32] | type=2
    return g;
}
// Build D# group 1 for a 2D bf16 tile: data_size=1 (2 bytes),
// tensor_dim0/1, tile_dim0/1, tensor_dim0_stride; no multicast/pad/iterate.
__device__ __forceinline__ i32x8 tdm_g1_2d(unsigned tdim0, unsigned tdim1,
                                           unsigned tile0, unsigned tile1,
                                           unsigned stride0) {
    i32x8 g = {};
    g[0] = 0x00010000;                                           // data_size = 2B
    g[1] = (int)((tdim0 & 0xFFFFu) << 16);                       // dim0[15:0] @ bits63:48
    g[2] = (int)(((tdim0 >> 16) & 0xFFFFu) | ((tdim1 & 0xFFFFu) << 16));
    g[3] = (int)(((tdim1 >> 16) & 0xFFFFu) | ((tile0 & 0xFFFFu) << 16));
    g[4] = (int)(tile1 & 0xFFFFu);                               // tile_dim2 = 0
    g[5] = (int)stride0;                                         // dim0_stride[31:0]
    g[6] = 0;                                                    // stride hi / dim1_stride
    g[7] = 0;
    return g;
}
__device__ __forceinline__ void tdm_load(unsigned lds_addr, unsigned long long gaddr,
                                         unsigned tdim0, unsigned tdim1,
                                         unsigned tile0, unsigned tile1,
                                         unsigned stride0) {
    const u32x4 g0 = tdm_g0(lds_addr, gaddr);
    const i32x8 g1 = tdm_g1_2d(tdim0, tdim1, tile0, tile1, stride0);
    const i32x4 z4 = {};
#if __clang_major__ >= 23
    const i32x8 z8 = {};
    __builtin_amdgcn_tensor_load_to_lds(g0, g1, z4, z4, z8, 0);
#else
    __builtin_amdgcn_tensor_load_to_lds(g0, g1, z4, z4, 0);
#endif
}
// Low 32 bits of a flat LDS pointer are the LDS byte offset (aperture rule).
__device__ __forceinline__ unsigned lds_off(const void* p) {
    return (unsigned)(uintptr_t)p;
}
#endif // MBD_USE_TDM

// -------------------------------------------------------------------------
// Prepass A: x [256][1024] f32 -> xh/xl bf16 row-major (packed pairs).
// -------------------------------------------------------------------------
__global__ __launch_bounds__(256) void mbd_cvt_x(
    const float* __restrict__ x,
    unsigned* __restrict__ xh, unsigned* __restrict__ xl)
{
    const int t = blockIdx.x * 256 + threadIdx.x;     // 0 .. 131071 pairs
    const float2 v = ((const float2*)x)[t];
    const unsigned h0 = f2bf(v.x), h1 = f2bf(v.y);
    xh[t] = h0 | (h1 << 16);
    const unsigned l0 = f2bf(v.x - bf2f(h0));
    const unsigned l1 = f2bf(v.y - bf2f(h1));
    xl[t] = l0 | (l1 << 16);
}

// -------------------------------------------------------------------------
// Prepass B: T [1024][2048] f32 -> Th/Tl bf16 TRANSPOSED [2048][1024].
// 32x32 LDS tile transpose; both global read and write coalesced.
// -------------------------------------------------------------------------
__global__ __launch_bounds__(256) void mbd_cvt_T(
    const float* __restrict__ T,
    unsigned* __restrict__ Th, unsigned* __restrict__ Tl)
{
    __shared__ float tile[32][33];
    const int n0 = blockIdx.x * 32;   // T column tile
    const int k0 = blockIdx.y * 32;   // T row (K) tile
    const int tid = threadIdx.x;

    for (int t = tid; t < 1024; t += 256) {
        const int kk = t >> 5, nn = t & 31;
        tile[kk][nn] = T[(k0 + kk) * BC + n0 + nn];
    }
    __syncthreads();
    for (int t = tid; t < 512; t += 256) {
        const int n = t >> 4, q = t & 15;
        const float v0 = tile[2 * q][n], v1 = tile[2 * q + 1][n];
        const unsigned h0 = f2bf(v0), h1 = f2bf(v1);
        Th[(size_t)(n0 + n) * 512 + (k0 >> 1) + q] = h0 | (h1 << 16);
        const unsigned l0 = f2bf(v0 - bf2f(h0));
        const unsigned l1 = f2bf(v1 - bf2f(h1));
        Tl[(size_t)(n0 + n) * 512 + (k0 >> 1) + q] = l0 | (l1 << 16);
    }
}

// -------------------------------------------------------------------------
// Kernel 1: m = x @ T via bf16 WMMA with hi/lo split (~tf32 accuracy).
// Tile staging via the Tensor Data Mover (tensor_load_to_lds, TENSORcnt),
// issued by wave 0; fragment loads are contiguous ds_load_b128.
// -------------------------------------------------------------------------
__global__ __launch_bounds__(256) void mbd_gemm_wmma(
    const unsigned* __restrict__ xh, const unsigned* __restrict__ xl,
    const unsigned* __restrict__ Th, const unsigned* __restrict__ Tl,
    float* __restrict__ m_t)
{
    __shared__ __align__(16) unsigned short sAh[16 * 32];   // 1 KB  [row][k]
    __shared__ __align__(16) unsigned short sAl[16 * 32];
    __shared__ __align__(16) unsigned short sBh[128 * 32];  // 8 KB  [col][k]
    __shared__ __align__(16) unsigned short sBl[128 * 32];

    const int colGroup = blockIdx.x;   // 0..15  (128 cols each)
    const int rowTile  = blockIdx.y;   // 0..15  (16 rows each)
    const int tid  = threadIdx.x;
    const int wave = tid >> 5;         // 0..7
    const int lane = tid & 31;

    const int row0    = rowTile * 16;
    const int colBase = colGroup * 128;
    const int col0    = colBase + wave * 16;

    const int mrow  = lane & 15;            // A: M index / B: N index
    const int khalf = (lane >> 4) * 8;      // K sub-block select per lane half
    const int bcol  = wave * 16 + mrow;     // B column within block tile

    v8f acc = {};

    for (int k0 = 0; k0 < ADIM; k0 += 32) {
        __syncthreads();   // all waves' fragment reads of prev step are done
#ifdef MBD_USE_TDM
        if (wave == 0) {
            const unsigned long long xhB = (unsigned long long)(uintptr_t)xh;
            const unsigned long long xlB = (unsigned long long)(uintptr_t)xl;
            const unsigned long long ThB = (unsigned long long)(uintptr_t)Th;
            const unsigned long long TlB = (unsigned long long)(uintptr_t)Tl;
            const unsigned long long aOff = ((unsigned long long)row0 * ADIM + k0) * 2ull;
            const unsigned long long bOff = ((unsigned long long)colBase * ADIM + k0) * 2ull;
            // A tiles: 16 rows x 32 K of bf16 from [256][1024]
            tdm_load(lds_off(sAh), xhB + aOff, ADIM, BATCH, 32, 16, ADIM);
            tdm_load(lds_off(sAl), xlB + aOff, ADIM, BATCH, 32, 16, ADIM);
            // B tiles: 128 cols x 32 K of bf16 from transposed [2048][1024]
            tdm_load(lds_off(sBh), ThB + bOff, ADIM, BC, 32, 128, ADIM);
            tdm_load(lds_off(sBl), TlB + bOff, ADIM, BC, 32, 128, ADIM);
            __builtin_amdgcn_s_wait_tensorcnt(0);
        }
#else
        // Fallback: cooperative VMEM->VGPR->LDS staging.
        {
            const int r = tid >> 4, q = tid & 15;
            const size_t ga = (size_t)(row0 + r) * 512 + (k0 >> 1) + q;
            ((unsigned*)sAh)[tid] = xh[ga];
            ((unsigned*)sAl)[tid] = xl[ga];
        }
        for (int t = tid; t < 512; t += 256) {
            const int cc = t >> 2, q = t & 3;
            const size_t gb = (size_t)(colBase + cc) * 512 + (k0 >> 1) + 4 * q;
            ((v4u*)sBh)[t] = *(const v4u*)&Th[gb];
            ((v4u*)sBl)[t] = *(const v4u*)&Tl[gb];
        }
#endif
        // Prefetch next K-step tiles into cache (global_prefetch_b8).
        if (k0 + 32 < ADIM) {
            __builtin_prefetch(
                &Th[(size_t)(colBase + (tid >> 1)) * 512 + ((k0 + 32) >> 1) + (tid & 1) * 8], 0, 2);
            if (tid < 16)
                __builtin_prefetch(&xh[(size_t)(row0 + tid) * 512 + ((k0 + 32) >> 1)], 0, 2);
        }
        __syncthreads();   // staged tiles visible to all waves

        FragU ah, al, bh, bl;
        ah.q[0] = *(const v4u*)&sAh[mrow * 32 + khalf];
        ah.q[1] = *(const v4u*)&sAh[mrow * 32 + 16 + khalf];
        al.q[0] = *(const v4u*)&sAl[mrow * 32 + khalf];
        al.q[1] = *(const v4u*)&sAl[mrow * 32 + 16 + khalf];
        bh.q[0] = *(const v4u*)&sBh[bcol * 32 + khalf];
        bh.q[1] = *(const v4u*)&sBh[bcol * 32 + 16 + khalf];
        bl.q[0] = *(const v4u*)&sBl[bcol * 32 + khalf];
        bl.q[1] = *(const v4u*)&sBl[bcol * 32 + 16 + khalf];

        acc = __builtin_amdgcn_wmma_f32_16x16x32_bf16(
            false, ah.b, false, bh.b, (short)0, acc, false, false);
        acc = __builtin_amdgcn_wmma_f32_16x16x32_bf16(
            false, ah.b, false, bl.b, (short)0, acc, false, false);
        acc = __builtin_amdgcn_wmma_f32_16x16x32_bf16(
            false, al.b, false, bh.b, (short)0, acc, false, false);
    }

    // C/D layout: VGPR r -> M = r + 8*(lane>=16), N = lane&15
    const int M = row0 + ((lane >> 4) << 3);
    const int N = col0 + (lane & 15);
    const int b = N >> 5;
    const int c = N & 31;
    #pragma unroll
    for (int r = 0; r < 8; ++r) {
        m_t[(size_t)b * (BATCH * CDIM) + (size_t)(M + r) * CDIM + c] = acc[r];
    }
}

// -------------------------------------------------------------------------
// Kernel 2: pairwise L1 + exp reduction. One block per b; m[b] (32 KB) in LDS.
// -------------------------------------------------------------------------
__global__ __launch_bounds__(256) void mbd_pairwise(
    const float* __restrict__ m_t,  // [64][256][32]
    float* __restrict__ out)        // [256][1088]
{
    __shared__ float sM[BATCH * CDIM];   // 32 KB

    const int b   = blockIdx.x;
    const int tid = threadIdx.x;         // i = tid
    const float* mb = m_t + (size_t)b * (BATCH * CDIM);

    for (int t = tid; t < (BATCH * CDIM) / 4; t += 256) {
        ((float4*)sM)[t] = ((const float4*)mb)[t];
    }
    __syncthreads();

    float mi[CDIM];
    #pragma unroll
    for (int c = 0; c < CDIM; ++c) mi[c] = sM[tid * CDIM + c];

    float o = 0.0f;
    for (int j = 0; j < BATCH; ++j) {
        const float* mj = &sM[j * CDIM];
        float s = 0.0f;
        #pragma unroll
        for (int c = 0; c < CDIM; ++c) s += fabsf(mi[c] - mj[c]);
        // the i==j pair is skipped by the select and adds exactly 0
        o += (j == tid) ? 0.0f : __expf(-s);
    }
    out[(size_t)tid * OUTW + ADIM + b] = o;
}

// -------------------------------------------------------------------------
// Kernel 3: copy x into out[:, 0:1024], float4 coalesced.
// -------------------------------------------------------------------------
__global__ __launch_bounds__(256) void mbd_copy_x(
    const float* __restrict__ x, float* __restrict__ out)
{
    const int t = blockIdx.x * 256 + threadIdx.x;  // float4 index
    const int i = t >> 8;
    const int a = t & 255;
    ((float4*)(out + (size_t)i * OUTW))[a] = ((const float4*)(x + (size_t)i * ADIM))[a];
}

extern "C" void kernel_launch(void* const* d_in, const int* in_sizes, int n_in,
                              void* d_out, int out_size, void* d_ws, size_t ws_size,
                              hipStream_t stream) {
    const float* x = (const float*)d_in[0];
    const float* T = (const float*)d_in[1];
    float* out = (float*)d_out;

    // Workspace: m_t 2MB | xh 0.5MB | xl 0.5MB | Th 4MB | Tl 4MB  (~11MB)
    char* ws = (char*)d_ws;
    float*    m_t = (float*)ws;                        ws += (size_t)BDIM * BATCH * CDIM * 4;
    unsigned* xh  = (unsigned*)ws;                     ws += (size_t)BATCH * ADIM * 2;
    unsigned* xl  = (unsigned*)ws;                     ws += (size_t)BATCH * ADIM * 2;
    unsigned* Th  = (unsigned*)ws;                     ws += (size_t)ADIM * BC * 2;
    unsigned* Tl  = (unsigned*)ws;

    mbd_cvt_x<<<(BATCH * ADIM / 2) / 256, 256, 0, stream>>>(x, xh, xl);
    mbd_cvt_T<<<dim3(BC / 32, ADIM / 32), 256, 0, stream>>>(T, Th, Tl);
    mbd_gemm_wmma<<<dim3(16, 16), 256, 0, stream>>>(xh, xl, Th, Tl, m_t);
    mbd_pairwise<<<BDIM, 256, 0, stream>>>(m_t, out);
    mbd_copy_x<<<256, 256, 0, stream>>>(x, out);
}
